// GroupedQueryAttentionV2_17927193493794
// MI455X (gfx1250) — compile-verified, matrix operand order checked by reading, not run on hardware
//
#include <hip/hip_runtime.h>

// ---------------------------------------------------------------------------
// Types / helpers
// ---------------------------------------------------------------------------
typedef __attribute__((ext_vector_type(16))) __bf16 v16bf;
typedef __attribute__((ext_vector_type(8)))  float  v8f;

#define WMMA_BF16(a, b, c) \
  __builtin_amdgcn_wmma_f32_16x16x32_bf16(false, (a), false, (b), (short)0, (c), false, false)

__device__ __forceinline__ unsigned short f32_to_bf16(float f) {
  unsigned int u = __float_as_uint(f);
  u += 0x7FFFu + ((u >> 16) & 1u);   // round-to-nearest-even
  return (unsigned short)(u >> 16);
}

// row reductions across the 16-lane half (C/D layout stripes N over lanes 0-15 / 16-31)
__device__ __forceinline__ float rowmax16(float v) {
  v = fmaxf(v, __shfl_xor(v, 1, 32));
  v = fmaxf(v, __shfl_xor(v, 2, 32));
  v = fmaxf(v, __shfl_xor(v, 4, 32));
  v = fmaxf(v, __shfl_xor(v, 8, 32));
  return v;
}
__device__ __forceinline__ float rowsum16(float v) {
  v += __shfl_xor(v, 1, 32);
  v += __shfl_xor(v, 2, 32);
  v += __shfl_xor(v, 4, 32);
  v += __shfl_xor(v, 8, 32);
  return v;
}

// ---------------------------------------------------------------------------
// Problem constants
// ---------------------------------------------------------------------------
#define CB   2
#define CS   2048
#define CH   2048
#define CNH  32
#define CNKV 8
#define CHD  64
#define CDQ  (CNH * CHD)    // 2048
#define CDKV (CNKV * CHD)   // 512
#define CM   (CB * CS)      // 4096

// ---------------------------------------------------------------------------
// fp32 -> bf16 elementwise convert
// ---------------------------------------------------------------------------
__global__ void cvt_bf16_kernel(const float* __restrict__ in,
                                unsigned short* __restrict__ out, int n) {
  int i = blockIdx.x * blockDim.x + threadIdx.x;
  if (i < n) out[i] = f32_to_bf16(in[i]);
}

// ---------------------------------------------------------------------------
// GEMM: C[M,N] = A[M,K](bf16, row-major) x W[N,K](bf16, row-major)^T
// block = 256 threads (8 waves); wave computes a 16(M) x 64(N) tile.
// grid = (N/512, M/16)
// ---------------------------------------------------------------------------
template <bool BF16OUT>
__global__ __launch_bounds__(256) void gemm_xwt_kernel(
    const unsigned short* __restrict__ A, const unsigned short* __restrict__ W,
    void* __restrict__ Cv, int M, int N, int K) {
  const int lane = threadIdx.x & 31;
  const int w    = threadIdx.x >> 5;
  const int m0   = blockIdx.y * 16;
  const int n0   = (blockIdx.x * 8 + w) * 64;
  const int mrow = lane & 15;            // A: row within tile / B: column within tile
  const int koff = (lane >> 4) * 16;     // K chunk per lane-half

  const unsigned short* arow = A + (size_t)(m0 + mrow) * K + koff;

  v8f acc[4] = {};
  for (int k0 = 0; k0 < K; k0 += 32) {
    v16bf a = *(const v16bf*)(arow + k0);
#pragma unroll
    for (int t = 0; t < 4; ++t) {
      const unsigned short* wr = W + (size_t)(n0 + t * 16 + mrow) * K + koff + k0;
      v16bf b = *(const v16bf*)wr;
      acc[t] = WMMA_BF16(a, b, acc[t]);
    }
  }

  const int Mbase = 8 * (lane >> 4);
  if constexpr (BF16OUT) {
    unsigned short* C = (unsigned short*)Cv;
#pragma unroll
    for (int t = 0; t < 4; ++t)
#pragma unroll
      for (int r = 0; r < 8; ++r)
        C[(size_t)(m0 + Mbase + r) * N + n0 + t * 16 + mrow] = f32_to_bf16(acc[t][r]);
  } else {
    float* C = (float*)Cv;
#pragma unroll
    for (int t = 0; t < 4; ++t)
#pragma unroll
      for (int r = 0; r < 8; ++r)
        C[(size_t)(m0 + Mbase + r) * N + n0 + t * 16 + mrow] = acc[t][r];
  }
}

// ---------------------------------------------------------------------------
// V transpose: V[b,s,kv*64+d] -> Vt[b,kv,d,s]  (all bf16 bits)
// ---------------------------------------------------------------------------
__global__ void transpose_v_kernel(const unsigned short* __restrict__ V,
                                   unsigned short* __restrict__ Vt) {
  int i = blockIdx.x * blockDim.x + threadIdx.x;
  if (i >= CM * CDKV) return;
  int d  = i % CHD;
  int kv = (i / CHD) % CNKV;
  int s  = (i / CDKV) % CS;
  int b  = i / (CDKV * CS);
  Vt[(((size_t)b * CNKV + kv) * CHD + d) * CS + s] = V[i];
}

// ---------------------------------------------------------------------------
// Flash attention, causal, GQA (q head h pairs with kv head h/4).
// block = 128 threads (4 waves); each wave owns a 16-row q tile.
// grid = (S/64, B*NH)
// ---------------------------------------------------------------------------
__global__ __launch_bounds__(128) void attn_kernel(
    const unsigned short* __restrict__ Q,   // [B*S, 2048]
    const unsigned short* __restrict__ Km,  // [B*S, 512]
    const unsigned short* __restrict__ Vt,  // [B, NKV, 64, S]
    unsigned short* __restrict__ O) {       // [B*S, 2048]
  __shared__ __align__(64) unsigned short ldsP[4][16 * 32];

  const int lane = threadIdx.x & 31;
  const int w    = threadIdx.x >> 5;
  const int qt   = blockIdx.x * 4 + w;
  const int q0   = qt * 16;
  const int head = blockIdx.y;
  const int b    = head >> 5;       // / NH
  const int hh   = head & 31;       // % NH
  const int kv   = hh >> 2;         // / NREP

  const unsigned short* Qb = Q  + (size_t)b * CS * CDQ  + (size_t)hh * CHD;
  const unsigned short* Kb = Km + (size_t)b * CS * CDKV + (size_t)kv * CHD;
  const unsigned short* Vb = Vt + (size_t)(b * CNKV + kv) * CHD * CS;
  unsigned short*       Ob = O  + (size_t)b * CS * CDQ  + (size_t)hh * CHD;

  const int mrow = lane & 15;
  const int ncol = lane & 15;
  const int koff = (lane >> 4) * 16;
  const int qrow_lane = q0 + 8 * (lane >> 4);  // + r gives this lane's row

  // Q A-fragments for d=0..31 and d=32..63
  v16bf aq0 = *(const v16bf*)(Qb + (size_t)(q0 + mrow) * CDQ + koff);
  v16bf aq1 = *(const v16bf*)(Qb + (size_t)(q0 + mrow) * CDQ + 32 + koff);

  float mr[8], lr[8];
  v8f acc[4] = {};
#pragma unroll
  for (int r = 0; r < 8; ++r) { mr[r] = -3.0e38f; lr[r] = 0.0f; }

  unsigned short* pl = &ldsP[w][0];

  for (int kv0 = 0; kv0 < q0 + 16; kv0 += 32) {
    // ---- scores: 16(q) x 32(kv), two 16x16 WMMA column tiles, K=64 chained
    v8f s[2];
#pragma unroll
    for (int j = 0; j < 2; ++j) {
      const int kvr = kv0 + j * 16 + ncol;
      v16bf bk0 = *(const v16bf*)(Kb + (size_t)kvr * CDKV + koff);
      v16bf bk1 = *(const v16bf*)(Kb + (size_t)kvr * CDKV + 32 + koff);
      v8f c = {};
      c = WMMA_BF16(aq0, bk0, c);
      c = WMMA_BF16(aq1, bk1, c);
      s[j] = c;
    }
    // ---- scale + causal mask
#pragma unroll
    for (int j = 0; j < 2; ++j) {
      const int kcol = kv0 + j * 16 + ncol;
#pragma unroll
      for (int r = 0; r < 8; ++r) {
        float v = s[j][r] * 0.125f;                 // HD^-0.5
        v += (kcol > qrow_lane + r) ? -1.0e9f : 0.0f;
        s[j][r] = v;
      }
    }
    // ---- online softmax stats
    float mnew[8], alpha[8];
#pragma unroll
    for (int r = 0; r < 8; ++r) {
      float v = rowmax16(fmaxf(s[0][r], s[1][r]));
      mnew[r]  = fmaxf(mr[r], v);
      alpha[r] = __expf(mr[r] - mnew[r]);
      mr[r]    = mnew[r];
    }
    float p0[8], p1[8];
#pragma unroll
    for (int r = 0; r < 8; ++r) {
      p0[r] = __expf(s[0][r] - mnew[r]);
      p1[r] = __expf(s[1][r] - mnew[r]);
      float v = rowsum16(p0[r] + p1[r]);
      lr[r] = lr[r] * alpha[r] + v;
    }
#pragma unroll
    for (int t = 0; t < 4; ++t)
#pragma unroll
      for (int r = 0; r < 8; ++r) acc[t][r] *= alpha[r];

    // ---- P (16x32 bf16) through per-wave LDS to convert C-layout -> A-fragment
#pragma unroll
    for (int r = 0; r < 8; ++r) {
      const int row = r + 8 * (lane >> 4);
      pl[row * 32 + ncol]      = f32_to_bf16(p0[r]);
      pl[row * 32 + 16 + ncol] = f32_to_bf16(p1[r]);
    }
    v16bf ap = *(const v16bf*)(pl + mrow * 32 + koff);

    // ---- acc += P x V   (V^T gives contiguous B-fragment loads)
#pragma unroll
    for (int t = 0; t < 4; ++t) {
      v16bf bv = *(const v16bf*)(Vb + (size_t)(t * 16 + ncol) * CS + kv0 + koff);
      acc[t] = WMMA_BF16(ap, bv, acc[t]);
    }
  }

  // ---- epilogue: divide by row sums, store bf16 O
  float inv[8];
#pragma unroll
  for (int r = 0; r < 8; ++r) inv[r] = 1.0f / lr[r];
#pragma unroll
  for (int t = 0; t < 4; ++t)
#pragma unroll
    for (int r = 0; r < 8; ++r) {
      const int row = q0 + r + 8 * (lane >> 4);
      Ob[(size_t)row * CDQ + t * 16 + ncol] = f32_to_bf16(acc[t][r] * inv[r]);
    }
}

// ---------------------------------------------------------------------------
// Host launcher
// ---------------------------------------------------------------------------
extern "C" void kernel_launch(void* const* d_in, const int* in_sizes, int n_in,
                              void* d_out, int out_size, void* d_ws, size_t ws_size,
                              hipStream_t stream) {
  (void)in_sizes; (void)n_in; (void)out_size; (void)ws_size;
  const float* hs = (const float*)d_in[0];
  // d_in[1] = attention_mask: causal, reproduced analytically in-kernel
  const float* Wq = (const float*)d_in[2];
  const float* Wk = (const float*)d_in[3];
  const float* Wv = (const float*)d_in[4];
  const float* Wo = (const float*)d_in[5];

  unsigned char* ws = (unsigned char*)d_ws;
  size_t off = 0;
  unsigned short* Xb   = (unsigned short*)(ws + off); off += (size_t)CM * CH * 2;     // 16MB (reused as O)
  unsigned short* Wbuf = (unsigned short*)(ws + off); off += (size_t)CH * CH * 2;     // 8MB shared weight buf
  unsigned short* Qb   = (unsigned short*)(ws + off); off += (size_t)CM * CDQ * 2;    // 16MB
  unsigned short* Kb   = (unsigned short*)(ws + off); off += (size_t)CM * CDKV * 2;   // 4MB
  unsigned short* Vb   = (unsigned short*)(ws + off); off += (size_t)CM * CDKV * 2;   // 4MB
  unsigned short* Vtb  = (unsigned short*)(ws + off); off += (size_t)CM * CDKV * 2;   // 4MB
  unsigned short* Ob   = Xb;  // X dead after V projection

  auto cvt = [&](const float* src, unsigned short* dst, int n) {
    cvt_bf16_kernel<<<(n + 255) / 256, 256, 0, stream>>>(src, dst, n);
  };

  cvt(hs, Xb, CM * CH);

  // Q = X Wq^T  [4096 x 2048 x 2048]
  cvt(Wq, Wbuf, CDQ * CH);
  gemm_xwt_kernel<true><<<dim3(CDQ / 512, CM / 16), 256, 0, stream>>>(Xb, Wbuf, Qb, CM, CDQ, CH);
  // K = X Wk^T  [4096 x 512 x 2048]
  cvt(Wk, Wbuf, CDKV * CH);
  gemm_xwt_kernel<true><<<dim3(CDKV / 512, CM / 16), 256, 0, stream>>>(Xb, Wbuf, Kb, CM, CDKV, CH);
  // V = X Wv^T
  cvt(Wv, Wbuf, CDKV * CH);
  gemm_xwt_kernel<true><<<dim3(CDKV / 512, CM / 16), 256, 0, stream>>>(Xb, Wbuf, Vb, CM, CDKV, CH);

  transpose_v_kernel<<<(CM * CDKV + 255) / 256, 256, 0, stream>>>(Vb, Vtb);

  attn_kernel<<<dim3(CS / 64, CB * CNH), 128, 0, stream>>>(Qb, Kb, Vtb, Ob);

  // out = O Wo^T  [4096 x 2048 x 2048], fp32 stores into d_out
  cvt(Wo, Wbuf, CH * CDQ);
  gemm_xwt_kernel<false><<<dim3(CH / 512, CM / 16), 256, 0, stream>>>(Ob, Wbuf, d_out, CM, CH, CDQ);
}
